// MultiHeadAttention_82506321756607
// MI455X (gfx1250) — matrix operand order repaired and checked by reading, not compiled, not run on hardware
//
#include <hip/hip_runtime.h>
#include <hip/hip_bf16.h>

// ---------------------------------------------------------------------------
// MHA forward, B=8, S=1024, HID=1024, NH=16, HD=64 on gfx1250 (CDNA5).
//  - all matmuls: v_wmma_f32_16x16x32_bf16 (wave32)
//  - projections: LDS-staged 128x128 block tiles, TDM (tensor_load_to_lds)
//    double-buffered with LDS row padding for conflict-free ds_load reads.
//  - attention: fused flash kernel; one block owns one (b,h) x 128 q-rows,
//    K/V tiles TDM-staged into LDS (shared by 8 waves), transposed score
//    tiles feed the P*V WMMA A-fragment with zero cross-lane movement.
// ---------------------------------------------------------------------------

typedef __bf16 bf16_t;
typedef __attribute__((ext_vector_type(16))) __bf16 v16bf;
typedef __attribute__((ext_vector_type(8)))  __bf16 v8bf;
typedef __attribute__((ext_vector_type(8)))  float  v8f;
typedef __attribute__((ext_vector_type(4)))  float  v4f;
typedef unsigned int __attribute__((ext_vector_type(4))) u32x4;
typedef int          __attribute__((ext_vector_type(4))) i32x4;
typedef int          __attribute__((ext_vector_type(8))) i32x8;

#define B_   8
#define S_   1024
#define HID_ 1024
#define NH_  16
#define HD_  64

// GEMM block tiling
#define BM 128
#define BN 128
#define BK 32
#define LDS_STRIDE 40   // 32 elems + 4-dword TDM pad -> 80B rows, conflict-free

// flash LDS tile strides
#define KSTR 72         // 64 elems + 4-dword pad -> 144B rows
#define VSTR 40         // 32 elems + 4-dword pad -> 80B rows

static __device__ __forceinline__ v8bf ld8(const bf16_t* p) {
    return *(const v8bf*)p;
}

// A fragment: 16x32 bf16 tile, rows r0.., cols c0.. of row-major src (stride ld).
// lanes 0-15: M=lane, K={0..7,16..23}; lanes 16-31: M=lane-16, K={8..15,24..31}
static __device__ __forceinline__ v16bf loadA(const bf16_t* src, int ld,
                                              int r0, int c0) {
    int lane = threadIdx.x & 31;
    int row  = r0 + (lane & 15);
    int cb   = c0 + ((lane >> 4) << 3);
    v8bf lo = ld8(src + (size_t)row * ld + cb);
    v8bf hi = ld8(src + (size_t)row * ld + cb + 16);
    v16bf a;
#pragma unroll
    for (int i = 0; i < 8; ++i) { a[i] = lo[i]; a[i + 8] = hi[i]; }
    return a;
}

// B fragment: 32x16 where B[k][n] = T[n][k] (T row-major, stride ld).
// lanes 0-15: N=lane, K={0..7,16..23}; lanes 16-31: N=lane-16, K={8..15,24..31}
static __device__ __forceinline__ v16bf loadBt(const bf16_t* T, int ld,
                                               int k0, int n0) {
    int lane = threadIdx.x & 31;
    int n  = n0 + (lane & 15);
    int kb = k0 + ((lane >> 4) << 3);
    v8bf lo = ld8(T + (size_t)n * ld + kb);
    v8bf hi = ld8(T + (size_t)n * ld + kb + 16);
    v16bf b;
#pragma unroll
    for (int i = 0; i < 8; ++i) { b[i] = lo[i]; b[i + 8] = hi[i]; }
    return b;
}

static __device__ __forceinline__ v8f wmma_bf16(v16bf a, v16bf b, v8f c) {
    return __builtin_amdgcn_wmma_f32_16x16x32_bf16(
        false, a, false, b, (short)0, c, false, false);
}

// ---------------------------------------------------------------------------
// TDM: load a 2D bf16 tile (tile_d0 x tile_d1, row stride d0_stride elems)
// from global into LDS, padding each row (pad_icode: 3 -> every 16 dwords,
// 4 -> every 32 dwords) with 4 dwords. D# packing per ISA 08_async_tensor §8.
// ---------------------------------------------------------------------------
static __device__ __forceinline__ void tdm_load_2d(unsigned lds_addr,
                                                   const bf16_t* gptr,
                                                   unsigned tensor_d0,
                                                   unsigned tensor_d1,
                                                   unsigned tile_d0,
                                                   unsigned tile_d1,
                                                   unsigned d0_stride,
                                                   unsigned pad_icode) {
    unsigned long long ga = (unsigned long long)(size_t)gptr;
    u32x4 g0;
    g0[0] = 1u;                                   // count=1, user descriptor
    g0[1] = lds_addr;                             // LDS byte address
    g0[2] = (unsigned)(ga & 0xffffffffu);         // global_addr[31:0]
    g0[3] = (unsigned)((ga >> 32) & 0x1ffffffu)   // global_addr[56:32]
          | (2u << 30);                           // type = 2 ("image")
    i32x8 g1;
    g1[0] = (int)((1u << 16)                      // data_size = 2 bytes
          | (1u << 20)                            // pad_enable
          | (pad_icode << 22)                     // pad_interval
          | (3u << 25));                          // pad_amount: 4 dwords
    g1[1] = (int)((tensor_d0 & 0xffffu) << 16);               // dim0[15:0]
    g1[2] = (int)((tensor_d0 >> 16) | ((tensor_d1 & 0xffffu) << 16));
    g1[3] = (int)((tensor_d1 >> 16) | (tile_d0 << 16));       // tile_dim0
    g1[4] = (int)(tile_d1 & 0xffffu);                         // tile_dim1
    g1[5] = (int)d0_stride;                                   // dim0 stride
    g1[6] = 0;
    g1[7] = 0;
    i32x4 z4 = {0, 0, 0, 0};
#if __clang_major__ >= 23
    i32x8 z8 = {0, 0, 0, 0, 0, 0, 0, 0};
    __builtin_amdgcn_tensor_load_to_lds(g0, g1, z4, z4, z8, 0);
#else
    __builtin_amdgcn_tensor_load_to_lds(g0, g1, z4, z4, 0);
#endif
}

static __device__ __forceinline__ unsigned lds_off(const void* p) {
    return (unsigned)(size_t)p;   // LDS aperture: low 32 bits = LDS byte addr
}

// ---------------------------------------------------------------------------
// fp32 -> bf16 bulk convert (x)
// ---------------------------------------------------------------------------
__global__ __launch_bounds__(256)
void cvt_bf16_kernel(const float* __restrict__ in, bf16_t* __restrict__ out, int n) {
    int i = (blockIdx.x * blockDim.x + threadIdx.x) * 4;
    if (i + 3 < n) {
        float4 f = *(const float4*)(in + i);
        out[i]     = (bf16_t)f.x;
        out[i + 1] = (bf16_t)f.y;
        out[i + 2] = (bf16_t)f.z;
        out[i + 3] = (bf16_t)f.w;
    }
}

// ---------------------------------------------------------------------------
// Transpose + convert 1024x1024 fp32 weight -> bf16 W^T (LDS tile)
// ---------------------------------------------------------------------------
__global__ __launch_bounds__(256)
void transpose_cvt_kernel(const float* __restrict__ in, bf16_t* __restrict__ out) {
    __shared__ float tile[16][17];
    int tx = threadIdx.x, ty = threadIdx.y;
    int i = blockIdx.y * 16 + ty;
    int j = blockIdx.x * 16 + tx;
    tile[ty][tx] = in[(size_t)i * HID_ + j];
    __syncthreads();
    out[(size_t)(blockIdx.x * 16 + ty) * HID_ + blockIdx.y * 16 + tx] =
        (bf16_t)tile[tx][ty];
}

// ---------------------------------------------------------------------------
// LDS-staged WMMA GEMM, TDM double-buffered.
// Block: 256 threads = 8 waves (2 in M x 4 in N), block tile 128x128,
// wave tile 64x32 (4x2 WMMA tiles), K step 32.
// mode 0/1: Q/K -> bf16 [BH][S][HD];  mode 2: V -> bf16 [BH][HD][S];
// mode 3: out projection -> fp32 [M][HID] + bias (writes d_out)
// ---------------------------------------------------------------------------
__global__ __launch_bounds__(256)
void proj_kernel(const bf16_t* __restrict__ X, const bf16_t* __restrict__ Wt,
                 const float* __restrict__ bias, void* __restrict__ outp,
                 int mode) {
    __shared__ bf16_t shA[2][BM * LDS_STRIDE];
    __shared__ bf16_t shB[2][BN * LDS_STRIDE];

    int wid  = threadIdx.x >> 5;
    int lane = threadIdx.x & 31;
    int wm = wid & 1;
    int wn = wid >> 1;

    const int nbn = HID_ / BN;   // 8
    int bm = blockIdx.x / nbn;
    int bn = blockIdx.x % nbn;
    int r0 = bm * BM, c0 = bn * BN;

    const bf16_t* Abase = X  + (size_t)r0 * HID_;
    const bf16_t* Bbase = Wt + (size_t)c0 * HID_;

    v8f acc[4][2] = {};

    if (wid == 0) {
        tdm_load_2d(lds_off(&shA[0][0]), Abase, HID_, BM, BK, BM, HID_, 3);
        tdm_load_2d(lds_off(&shB[0][0]), Bbase, HID_, BN, BK, BN, HID_, 3);
        __builtin_amdgcn_s_wait_tensorcnt(0);
    }
    __syncthreads();

    int buf = 0;
    for (int k0 = 0; k0 < HID_; k0 += BK) {
        if (wid == 0 && k0 + BK < HID_) {
            tdm_load_2d(lds_off(&shA[buf ^ 1][0]), Abase + (k0 + BK),
                        HID_, BM, BK, BM, HID_, 3);
            tdm_load_2d(lds_off(&shB[buf ^ 1][0]), Bbase + (k0 + BK),
                        HID_, BN, BK, BN, HID_, 3);
        }

        v16bf af[4];
#pragma unroll
        for (int i = 0; i < 4; ++i)
            af[i] = loadA(&shA[buf][0], LDS_STRIDE, wm * 64 + i * 16, 0);
#pragma unroll
        for (int j = 0; j < 2; ++j) {
            v16bf bf = loadBt(&shB[buf][0], LDS_STRIDE, 0, wn * 32 + j * 16);
#pragma unroll
            for (int i = 0; i < 4; ++i)
                acc[i][j] = wmma_bf16(af[i], bf, acc[i][j]);
        }

        if (wid == 0 && k0 + BK < HID_)
            __builtin_amdgcn_s_wait_tensorcnt(0);
        __syncthreads();
        buf ^= 1;
    }

    int half = lane >> 4;
#pragma unroll
    for (int i = 0; i < 4; ++i) {
#pragma unroll
        for (int j = 0; j < 2; ++j) {
            int col = c0 + wn * 32 + j * 16 + (lane & 15);
            float bb = bias[col];
#pragma unroll
            for (int vr = 0; vr < 8; ++vr) {
                int row = r0 + wm * 64 + i * 16 + vr + 8 * half;
                float val = acc[i][j][vr] + bb;
                if (mode == 3) {
                    ((float*)outp)[(size_t)row * HID_ + col] = val;
                } else {
                    int b = row >> 10, s = row & (S_ - 1);
                    int h = col >> 6, dd = col & (HD_ - 1);
                    int bh = b * NH_ + h;
                    bf16_t* o = (bf16_t*)outp;
                    if (mode == 2)
                        o[((size_t)bh * HD_ + dd) * S_ + s] = (bf16_t)val;
                    else
                        o[((size_t)bh * S_ + s) * HD_ + dd] = (bf16_t)val;
                }
            }
        }
    }
}

// ---------------------------------------------------------------------------
// Fused flash attention. Block = 8 waves, all sharing one (b,h); wave w owns
// q rows q0..q0+15. K (32x64) and V^T (64x32) tiles for the current 32-key
// chunk are TDM-staged into LDS, double-buffered, consumed by all 8 waves.
// Scores computed transposed: C[k,q] = K * Q^T; two consecutive 16-k score
// tiles, bf16-converted, ARE the A fragment of the 16x16x32 P*V WMMA.
// ---------------------------------------------------------------------------
__global__ __launch_bounds__(256)
void flash_kernel(const bf16_t* __restrict__ Q, const bf16_t* __restrict__ K,
                  const bf16_t* __restrict__ Vt, const float* __restrict__ mask,
                  bf16_t* __restrict__ ctx) {
    __shared__ bf16_t shK[2][32 * KSTR];   // 2 x 4.5 KB
    __shared__ bf16_t shV[2][64 * VSTR];   // 2 x 5   KB

    int wid  = threadIdx.x >> 5;
    int bh = blockIdx.x >> 3;                        // 8 blocks per (b,h)
    int q0 = ((blockIdx.x & 7) << 7) + (wid << 4);   // this wave's q strip
    int b = bh >> 4, h = bh & 15;

    const bf16_t* q  = Q  + (size_t)bh * S_ * HD_;
    const bf16_t* kp = K  + (size_t)bh * S_ * HD_;
    const bf16_t* vt = Vt + (size_t)bh * HD_ * S_;
    const float*  mk = mask + (size_t)b * S_ * S_;

    int lane = threadIdx.x & 31;
    int half = lane >> 4;
    int lq   = lane & 15;

    // Q strip in registers for the whole loop
    v16bf qB0 = loadBt(q, HD_, 0,  q0);
    v16bf qB1 = loadBt(q, HD_, 32, q0);

    // prologue: stage first K/V chunk
    if (wid == 0) {
        tdm_load_2d(lds_off(&shK[0][0]), kp, HD_, S_, HD_, 32, HD_, 4);
        tdm_load_2d(lds_off(&shV[0][0]), vt, S_, HD_, 32, HD_, S_, 3);
        __builtin_amdgcn_s_wait_tensorcnt(0);
    }
    __syncthreads();

    const float sc = 0.03125f;                   // 1/sqrt(HID)
    float m = -1e30f, l = 0.f;
    v8f accV[4] = {};

    int buf = 0;
    for (int kc = 0; kc < S_; kc += 32) {
        if (wid == 0 && kc + 32 < S_) {
            tdm_load_2d(lds_off(&shK[buf ^ 1][0]), kp + (size_t)(kc + 32) * HD_,
                        HD_, S_, HD_, 32, HD_, 4);
            tdm_load_2d(lds_off(&shV[buf ^ 1][0]), vt + (kc + 32),
                        S_, HD_, 32, HD_, S_, 3);
        }

        // transposed score tiles from LDS-staged K
        v8f s0 = {}, s1 = {};
        {
            v16bf a0 = loadA(&shK[buf][0], KSTR, 0,  0);
            v16bf a1 = loadA(&shK[buf][0], KSTR, 0,  32);
            s0 = wmma_bf16(a0, qB0, s0);
            s0 = wmma_bf16(a1, qB1, s0);
            v16bf a2 = loadA(&shK[buf][0], KSTR, 16, 0);
            v16bf a3 = loadA(&shK[buf][0], KSTR, 16, 32);
            s1 = wmma_bf16(a2, qB0, s1);
            s1 = wmma_bf16(a3, qB1, s1);
        }
        // scale + additive mask (per lane: fixed q, contiguous k)
        const float* mrow = mk + (size_t)(q0 + lq) * S_ + kc + 8 * half;
        v4f m0a = *(const v4f*)(mrow);
        v4f m0b = *(const v4f*)(mrow + 4);
        v4f m1a = *(const v4f*)(mrow + 16);
        v4f m1b = *(const v4f*)(mrow + 20);
#pragma unroll
        for (int r = 0; r < 4; ++r) {
            s0[r]     = s0[r]     * sc + m0a[r] * (-1e9f);
            s0[r + 4] = s0[r + 4] * sc + m0b[r] * (-1e9f);
            s1[r]     = s1[r]     * sc + m1a[r] * (-1e9f);
            s1[r + 4] = s1[r + 4] * sc + m1b[r] * (-1e9f);
        }
        // online softmax: lane pair (L, L^16) shares a q column
        float mx = -1e30f;
#pragma unroll
        for (int r = 0; r < 8; ++r) mx = fmaxf(mx, fmaxf(s0[r], s1[r]));
        mx = fmaxf(mx, __shfl_xor(mx, 16, 32));
        float mn = fmaxf(m, mx);
        float alpha = __expf(m - mn);

        float rs = 0.f;
        v16bf pA;
#pragma unroll
        for (int r = 0; r < 8; ++r) {
            float p0 = __expf(s0[r] - mn);
            float p1 = __expf(s1[r] - mn);
            rs += p0 + p1;
            pA[r]     = (bf16_t)p0;
            pA[r + 8] = (bf16_t)p1;
        }
        rs += __shfl_xor(rs, 16, 32);
        l = l * alpha + rs;
        m = mn;

        float af[8];
#pragma unroll
        for (int r = 0; r < 8; ++r) af[r] = __shfl(alpha, r + 8 * half, 32);
#pragma unroll
        for (int j = 0; j < 4; ++j) {
#pragma unroll
            for (int r = 0; r < 8; ++r) accV[j][r] *= af[r];
            v16bf bV = loadBt(&shV[buf][0], VSTR, 0, j * 16);
            accV[j] = wmma_bf16(pA, bV, accV[j]);
        }

        if (wid == 0 && kc + 32 < S_)
            __builtin_amdgcn_s_wait_tensorcnt(0);
        __syncthreads();
        buf ^= 1;
    }

    float rl = 1.f / l;
    float lf[8];
#pragma unroll
    for (int r = 0; r < 8; ++r) lf[r] = __shfl(rl, r + 8 * half, 32);
#pragma unroll
    for (int j = 0; j < 4; ++j) {
        int d = j * 16 + lq;
#pragma unroll
        for (int r = 0; r < 8; ++r) {
            int qrow = q0 + r + 8 * half;
            ctx[((size_t)(b * S_ + qrow)) * HID_ + h * HD_ + d] =
                (bf16_t)(accV[j][r] * lf[r]);
        }
    }
}

// ---------------------------------------------------------------------------
extern "C" void kernel_launch(void* const* d_in, const int* in_sizes, int n_in,
                              void* d_out, int out_size, void* d_ws, size_t ws_size,
                              hipStream_t stream) {
    const float* x   = (const float*)d_in[0];
    const float* msk = (const float*)d_in[1];
    const float* Wq  = (const float*)d_in[2];
    const float* bq  = (const float*)d_in[3];
    const float* Wk  = (const float*)d_in[4];
    const float* bk  = (const float*)d_in[5];
    const float* Wv  = (const float*)d_in[6];
    const float* bv  = (const float*)d_in[7];
    const float* Wp  = (const float*)d_in[8];
    const float* bp  = (const float*)d_in[9];

    char* ws = (char*)d_ws;
    size_t o = 0;
    bf16_t* xb  = (bf16_t*)(ws + o); o += (size_t)B_ * S_ * HID_ * 2;
    bf16_t* wqt = (bf16_t*)(ws + o); o += (size_t)HID_ * HID_ * 2;
    bf16_t* wkt = (bf16_t*)(ws + o); o += (size_t)HID_ * HID_ * 2;
    bf16_t* wvt = (bf16_t*)(ws + o); o += (size_t)HID_ * HID_ * 2;
    bf16_t* wpt = (bf16_t*)(ws + o); o += (size_t)HID_ * HID_ * 2;
    bf16_t* qb  = (bf16_t*)(ws + o); o += (size_t)B_ * NH_ * S_ * HD_ * 2;
    bf16_t* kb  = (bf16_t*)(ws + o); o += (size_t)B_ * NH_ * S_ * HD_ * 2;
    bf16_t* vtb = (bf16_t*)(ws + o); o += (size_t)B_ * NH_ * HD_ * S_ * 2;
    bf16_t* ctx = (bf16_t*)(ws + o); o += (size_t)B_ * S_ * HID_ * 2;
    (void)ws_size; (void)in_sizes; (void)n_in; (void)out_size;

    {   // x -> bf16
        int n = B_ * S_ * HID_;
        cvt_bf16_kernel<<<n / (256 * 4), 256, 0, stream>>>(x, xb, n);
    }
    {   // W -> W^T bf16
        dim3 g(HID_ / 16, HID_ / 16), t(16, 16);
        transpose_cvt_kernel<<<g, t, 0, stream>>>(Wq, wqt);
        transpose_cvt_kernel<<<g, t, 0, stream>>>(Wk, wkt);
        transpose_cvt_kernel<<<g, t, 0, stream>>>(Wv, wvt);
        transpose_cvt_kernel<<<g, t, 0, stream>>>(Wp, wpt);
    }
    {   // Q/K/V projections
        int blocks = (B_ * S_ / BM) * (HID_ / BN);   // 512
        proj_kernel<<<blocks, 256, 0, stream>>>(xb, wqt, bq, qb,  0);
        proj_kernel<<<blocks, 256, 0, stream>>>(xb, wkt, bk, kb,  1);
        proj_kernel<<<blocks, 256, 0, stream>>>(xb, wvt, bv, vtb, 2);
    }
    {   // fused attention: 128 (b,h) x 8 q-groups
        int blocks = B_ * NH_ * 8;                   // 1024
        flash_kernel<<<blocks, 256, 0, stream>>>(qb, kb, vtb, msk, ctx);
    }
    {   // output projection -> fp32 d_out
        int blocks = (B_ * S_ / BM) * (HID_ / BN);   // 512
        proj_kernel<<<blocks, 256, 0, stream>>>(ctx, wpt, bp, (void*)d_out, 3);
    }
}
